// XEyTransformerLayer_19713899889392
// MI455X (gfx1250) — compile-verified, matrix operand order checked by reading, not run on hardware
//
#include <hip/hip_runtime.h>
#include <hip/hip_bf16.h>
#include <math.h>

// ---------------------------------------------------------------------------
// XEyTransformerLayer forward, fused for MI455X (gfx1250, wave32, WMMA).
// All heavy GEMMs use v_wmma_f32_16x16x32_bf16. The E path is fully fused:
// E is read once and written once; Y / newE live only in LDS tiles; attention
// softmax over j is done online in registers (tile-granular rescaling), split
// across 2 j-halves per (b, i-tile) for 128-WG occupancy, merged by k_attfin.
// ---------------------------------------------------------------------------

typedef __attribute__((ext_vector_type(16))) __bf16 bf16x16;
typedef __attribute__((ext_vector_type(8)))  float  floatx8;

#define EPS_ 1e-5f

__device__ __forceinline__ unsigned short f2bf(float f) {
  unsigned u = __float_as_uint(f);
  u += 0x7FFFu + ((u >> 16) & 1u);           // round-to-nearest-even
  return (unsigned short)(u >> 16);
}

__device__ __forceinline__ unsigned packbf(float a, float b) {
  return (unsigned)f2bf(a) | ((unsigned)f2bf(b) << 16);
}

#define WMMA_BF(acc, a, b)                                                     \
  acc = __builtin_amdgcn_wmma_f32_16x16x32_bf16(false, (a), false, (b),        \
                                                (short)0, (acc), false, false)

// Load a 16x32 bf16 WMMA operand fragment from LDS stored row-major [rc][K].
// Lane L holds row/col rc0+(L&15); K halves per ISA layout:
//   ks = kb + (L>=16 ? 8 : 0); VGPR0..3 -> k=ks..ks+7; VGPR4..7 -> k=ks+16..ks+23
__device__ __forceinline__ bf16x16 frag_ld(const unsigned short* base, int rc0,
                                           int strideK, int kb) {
  int lane = threadIdx.x & 31;
  const unsigned short* p =
      base + (rc0 + (lane & 15)) * strideK + kb + ((lane & 16) ? 8 : 0);
  union { bf16x16 v; unsigned u[8]; } f;
#pragma unroll
  for (int i = 0; i < 4; ++i) f.u[i] = *(const unsigned*)(p + 2 * i);
#pragma unroll
  for (int i = 0; i < 4; ++i) f.u[4 + i] = *(const unsigned*)(p + 16 + 2 * i);
  return f.v;
}

// C/D layout: lane L -> col n0+(L&15); VGPR v -> row m0 + 8*(L>=16) + v.
__device__ __forceinline__ void cstore_f32(float* base, int m0, int n0,
                                           int strideN, floatx8 c) {
  int lane = threadIdx.x & 31;
  int n = n0 + (lane & 15);
  int m = m0 + ((lane & 16) ? 8 : 0);
#pragma unroll
  for (int v = 0; v < 8; ++v) base[(m + v) * strideN + n] = c[v];
}

__device__ __forceinline__ void cstore_relu_bf(unsigned short* base, int m0,
                                               int n0, int strideN, floatx8 c,
                                               const float* bias) {
  int lane = threadIdx.x & 31;
  int n = n0 + (lane & 15);
  int m = m0 + ((lane & 16) ? 8 : 0);
  float bv = bias[n];
#pragma unroll
  for (int v = 0; v < 8; ++v) {
    float t = fmaxf(c[v] + bv, 0.0f);
    base[(m + v) * strideN + n] = f2bf(t);
  }
}

// ---------------------------------------------------------------------------
// Kernel 1: Q/K/V projections. M=1024 rows, K=N=256. Q pre-scaled by 1/sqrt(32)
// and all three masked by node_mask (reference: (X@W+b)*xm).
// ---------------------------------------------------------------------------
__global__ void __launch_bounds__(256)
k_xproj(const float* __restrict__ X, const float* __restrict__ mask,
        const float* __restrict__ Wq, const float* __restrict__ bq,
        const float* __restrict__ Wk, const float* __restrict__ bk,
        const float* __restrict__ Wv, const float* __restrict__ bv,
        float* __restrict__ Qo, float* __restrict__ Ko, float* __restrict__ Vo) {
  extern __shared__ char smx[];
  unsigned short* sX = (unsigned short*)smx;           // [64][256] bf16
  unsigned short* sB = (unsigned short*)(smx + 32768); // [64][256] bf16 (B^T)
  float* sC = (float*)(smx + 65536);                   // [64][64]
  float* sM = (float*)(smx + 65536 + 16384);           // [64]
  int t = threadIdx.x, w = t >> 5;
  int rows0 = blockIdx.x * 64;

  for (int idx = t; idx < 64 * 256; idx += 256) {
    int r = idx >> 8, k = idx & 255;
    sX[r * 256 + k] = f2bf(X[(rows0 + r) * 256 + k]);
  }
  if (t < 64) sM[t] = mask[rows0 + t];
  __syncthreads();

  const float* Ws[3] = {Wq, Wk, Wv};
  const float* bs[3] = {bq, bk, bv};
  float* Ds[3] = {Qo, Ko, Vo};
  int ntv = w >> 1, mtb = (w & 1) * 2; // per-wave fixed nt, 2 m-tiles
  for (int m = 0; m < 3; ++m) {
    const float* W = Ws[m];
    const float* bias = bs[m];
    float* D = Ds[m];
    float scale = (m == 0) ? 0.17677669529663687f : 1.0f; // 1/sqrt(DF=32)
    for (int nch = 0; nch < 4; ++nch) {
      int n0 = nch * 64;
      for (int idx = t; idx < 64 * 256; idx += 256) {
        int n = idx & 63, k = idx >> 6;
        sB[n * 256 + k] = f2bf(W[k * 256 + n0 + n]); // transpose to [n][k]
      }
      __syncthreads();
      {
        floatx8 acc0 = {0.f, 0.f, 0.f, 0.f, 0.f, 0.f, 0.f, 0.f};
        floatx8 acc1 = {0.f, 0.f, 0.f, 0.f, 0.f, 0.f, 0.f, 0.f};
#pragma unroll
        for (int ks = 0; ks < 8; ++ks) {
          bf16x16 bb = frag_ld(sB, ntv * 16, 256, ks * 32);
          bf16x16 a0 = frag_ld(sX, mtb * 16, 256, ks * 32);
          bf16x16 a1 = frag_ld(sX, (mtb + 1) * 16, 256, ks * 32);
          WMMA_BF(acc0, a0, bb);
          WMMA_BF(acc1, a1, bb);
        }
        cstore_f32(sC, mtb * 16, ntv * 16, 64, acc0);
        cstore_f32(sC, (mtb + 1) * 16, ntv * 16, 64, acc1);
      }
      __syncthreads();
      for (int idx = t; idx < 64 * 64; idx += 256) {
        int r = idx >> 6, c = idx & 63;
        D[(rows0 + r) * 256 + n0 + c] =
            (sC[idx] + bias[n0 + c]) * sM[r] * scale;
      }
      __syncthreads();
    }
  }
}

// ---------------------------------------------------------------------------
// Kernel 2: per-batch FiLM vectors from y: y_e_add/mul, y_x_add/mul (8x256 each)
// ---------------------------------------------------------------------------
__global__ void k_ymats(const float* __restrict__ y,
                        const float* Wyea, const float* byea,
                        const float* Wyem, const float* byem,
                        const float* Wyxa, const float* byxa,
                        const float* Wyxm, const float* byxm,
                        float* oYEA, float* oYEM, float* oYXA, float* oYXM) {
  __shared__ float yr[64];
  int b = blockIdx.x, t = threadIdx.x;
  if (t < 64) yr[t] = y[b * 64 + t];
  __syncthreads();
  const float* Wm[4] = {Wyea, Wyem, Wyxa, Wyxm};
  const float* bm[4] = {byea, byem, byxa, byxm};
  float* om[4] = {oYEA, oYEM, oYXA, oYXM};
  for (int m = 0; m < 4; ++m) {
    float s = bm[m][t];
#pragma unroll 8
    for (int k = 0; k < 64; ++k) s += yr[k] * Wm[m][k * 256 + t];
    om[m][b * 256 + t] = s;
  }
}

// ---------------------------------------------------------------------------
// Kernel 3: the fused E path + online-softmax attention (partial over j-half).
// Grid: (i_tile=8, b=8, j_half=2) = 128 WGs. Block: 256 threads (8 wave32s).
// ---------------------------------------------------------------------------
#define EM_WMUL 0
#define EM_WADD 32768
#define EM_WEOUT 65536
#define EM_W1 98304
#define EM_W2 114688
#define EM_SQ 131072
#define EM_SK 147456
#define EM_SV 163840
#define EM_SA 180224
#define EM_SCR 212992   // eout/FF stage  float[256][64]
#define EM_SNE 212992   // ushort[256][32] (chunk phase)
#define EM_SE2 229376   // float[256][32]  (chunk phase)
#define EM_SH 278528    // float[256][32] Y / ushort[256][64] FF hidden
#define EM_BIAS 311296
#define EM_TOTAL 319616

__global__ void __launch_bounds__(256)
k_emain(const float* __restrict__ E, const float* __restrict__ mask,
        const float* __restrict__ Qs, const float* __restrict__ Kb,
        const float* __restrict__ Vb,
        const float* __restrict__ Wemul, const float* __restrict__ bemul,
        const float* __restrict__ Weadd, const float* __restrict__ beadd,
        const float* __restrict__ Weout, const float* __restrict__ beout,
        const float* __restrict__ W1, const float* __restrict__ b1,
        const float* __restrict__ W2, const float* __restrict__ b2,
        const float* __restrict__ ln1w, const float* __restrict__ ln1b,
        const float* __restrict__ ln2w, const float* __restrict__ ln2b,
        const float* __restrict__ yea, const float* __restrict__ yem,
        float* __restrict__ Eout,
        float* __restrict__ wsM, float* __restrict__ wsS,
        float* __restrict__ wsA) {
  extern __shared__ char sm[];
  unsigned short* sWmul = (unsigned short*)(sm + EM_WMUL);   // [256 n][64 k]
  unsigned short* sWadd = (unsigned short*)(sm + EM_WADD);
  unsigned short* sWeout = (unsigned short*)(sm + EM_WEOUT); // [64 n][256 k]
  unsigned short* sW1 = (unsigned short*)(sm + EM_W1);       // [128 n][64 k]
  unsigned short* sW2 = (unsigned short*)(sm + EM_W2);       // [64 n][128 k]
  float* sQ = (float*)(sm + EM_SQ);                          // [16][256]
  float* sK = (float*)(sm + EM_SK);
  float* sV = (float*)(sm + EM_SV);
  unsigned short* sA = (unsigned short*)(sm + EM_SA);        // [256 r][64 k]
  unsigned* sA32 = (unsigned*)(sm + EM_SA);
  unsigned short* sNE = (unsigned short*)(sm + EM_SNE);
  float* sE2f = (float*)(sm + EM_SE2);
  float* sScrF = (float*)(sm + EM_SCR);
  float* sHf = (float*)(sm + EM_SH);
  unsigned short* sHu = (unsigned short*)(sm + EM_SH);
  float* sB = (float*)(sm + EM_BIAS);
  float* bMul = sB;           float* bAdd = sB + 256;  float* bEo = sB + 512;
  float* B1 = sB + 576;       float* B2 = sB + 704;
  float* L1w = sB + 768;      float* L1b = sB + 832;
  float* L2w = sB + 896;      float* L2b = sB + 960;
  float* YEA = sB + 1024;     float* YEM = sB + 1280;
  float* MI = sB + 2048;      float* MJ = sB + 2064;

  int t = threadIdx.x, w = t >> 5;
  int i0 = blockIdx.x * 16, b = blockIdx.y, jh = blockIdx.z;

  // ---- one-time preload (weights transposed to [n][k] bf16) ----
  for (int idx = t; idx < 16384; idx += 256) {
    int n = idx & 255, k = idx >> 8;
    sWmul[n * 64 + k] = f2bf(Wemul[k * 256 + n]);
    sWadd[n * 64 + k] = f2bf(Weadd[k * 256 + n]);
  }
  for (int idx = t; idx < 16384; idx += 256) {
    int n = idx & 63, k = idx >> 6;
    sWeout[n * 256 + k] = f2bf(Weout[k * 64 + n]);
  }
  for (int idx = t; idx < 8192; idx += 256) {
    int n = idx & 127, k = idx >> 7;
    sW1[n * 64 + k] = f2bf(W1[k * 128 + n]);
  }
  for (int idx = t; idx < 8192; idx += 256) {
    int n = idx & 63, k = idx >> 6;
    sW2[n * 128 + k] = f2bf(W2[k * 64 + n]);
  }
  bMul[t] = bemul[t]; bAdd[t] = beadd[t];
  YEA[t] = yea[b * 256 + t]; YEM[t] = yem[b * 256 + t];
  if (t < 64) {
    bEo[t] = beout[t]; B2[t] = b2[t];
    L1w[t] = ln1w[t]; L1b[t] = ln1b[t]; L2w[t] = ln2w[t]; L2b[t] = ln2b[t];
  }
  if (t < 128) B1[t] = b1[t];
  if (t < 16) MI[t] = mask[b * 128 + i0 + t];
  {
    const float4* Q4 = (const float4*)Qs;
    float4* sQ4 = (float4*)sQ;
    for (int idx = t; idx < 1024; idx += 256) {
      int il = idx >> 6, c4 = idx & 63;
      sQ4[idx] = Q4[(b * 128 + i0 + il) * 64 + c4];
    }
  }

  // online softmax state: thread owns (i_local = t&15, c = (t>>4) + 16*slot)
  float smM[16], smS[16], smA[16];
#pragma unroll
  for (int k = 0; k < 16; ++k) { smM[k] = -1e30f; smS[k] = 0.f; smA[k] = 0.f; }
  int il_t = t & 15, cb_t = t >> 4;

  for (int jtl = 0; jtl < 4; ++jtl) {
    int j0 = jh * 64 + jtl * 16;
    __syncthreads();
    {
      const float4* K4 = (const float4*)Kb;
      const float4* V4 = (const float4*)Vb;
      float4* sK4 = (float4*)sK;
      float4* sV4 = (float4*)sV;
      for (int idx = t; idx < 1024; idx += 256) {
        int jl = idx >> 6, c4 = idx & 63;
        sK4[idx] = K4[(b * 128 + j0 + jl) * 64 + c4];
        sV4[idx] = V4[(b * 128 + j0 + jl) * 64 + c4];
      }
    }
    if (t < 16) MJ[t] = mask[b * 128 + j0 + t];
    int ri = i0 + (t >> 4), rj = j0 + (t & 15);
    const float* erow = E + ((size_t)(b * 128 + ri) * 128 + rj) * 64;
    float eR[64];
    {
      const float4* er4 = (const float4*)erow;
#pragma unroll
      for (int k4 = 0; k4 < 16; ++k4) {
        float4 v4 = er4[k4];
        eR[4 * k4 + 0] = v4.x; eR[4 * k4 + 1] = v4.y;
        eR[4 * k4 + 2] = v4.z; eR[4 * k4 + 3] = v4.w;
      }
#pragma unroll
      for (int m2 = 0; m2 < 32; ++m2)
        sA32[t * 32 + m2] = packbf(eR[2 * m2], eR[2 * m2 + 1]);
    }
    // software prefetch of the next j-tile's streams (global_prefetch_b8)
    if (jtl < 3) {
      __builtin_prefetch(erow + 1024, 0, 3);            // next E row (+16 j)
      __builtin_prefetch(erow + 1024 + 32, 0, 3);
      const float* nk = Kb + (size_t)(b * 128 + j0 + 16 + (t & 15)) * 256 +
                        (t >> 4) * 16;
      __builtin_prefetch(nk, 0, 3);
      const float* nv = Vb + (size_t)(b * 128 + j0 + 16 + (t & 15)) * 256 +
                        (t >> 4) * 16;
      __builtin_prefetch(nv, 0, 3);
    }
    __syncthreads();

    floatx8 eacc[8];
#pragma unroll
    for (int q = 0; q < 8; ++q)
      eacc[q] = (floatx8){0.f, 0.f, 0.f, 0.f, 0.f, 0.f, 0.f, 0.f};

#pragma unroll
    for (int nc = 0; nc < 8; ++nc) {
      int c0 = nc * 32;
      // --- E1/E2 projection GEMMs; per wave: fixed nt, B frags hoisted,
      //     A frag pair shared between the mul and add matrices ---
      {
        int ntv = w >> 2;          // 0..1
        int mtb = (w & 3) * 4;     // 0,4,8,12
        bf16x16 bm0 = frag_ld(sWmul, c0 + ntv * 16, 64, 0);
        bf16x16 bm1 = frag_ld(sWmul, c0 + ntv * 16, 64, 32);
        bf16x16 ba0 = frag_ld(sWadd, c0 + ntv * 16, 64, 0);
        bf16x16 ba1 = frag_ld(sWadd, c0 + ntv * 16, 64, 32);
#pragma unroll
        for (int p = 0; p < 4; ++p) {
          int mt = mtb + p;
          bf16x16 a0 = frag_ld(sA, mt * 16, 64, 0);
          bf16x16 a1 = frag_ld(sA, mt * 16, 64, 32);
          floatx8 accM = {0.f, 0.f, 0.f, 0.f, 0.f, 0.f, 0.f, 0.f};
          floatx8 accA = {0.f, 0.f, 0.f, 0.f, 0.f, 0.f, 0.f, 0.f};
          WMMA_BF(accM, a0, bm0);
          WMMA_BF(accM, a1, bm1);
          WMMA_BF(accA, a0, ba0);
          WMMA_BF(accA, a1, ba1);
          cstore_f32(sHf, mt * 16, ntv * 16, 32, accM);
          cstore_f32(sE2f, mt * 16, ntv * 16, 32, accA);
        }
      }
      __syncthreads();
      // --- Y = q*k*(em*E1+1) + em*E2 ; newE_pre = yea + (yem+1)*Y ---
#pragma unroll 4
      for (int ii = 0; ii < 32; ++ii) {
        int idx = t + 256 * ii;
        int r = idx >> 5, cc = idx & 31;
        int c = c0 + cc;
        float e1 = sHf[idx] + bMul[c];
        float e2 = sE2f[idx] + bAdd[c];
        int il = r >> 4, jl = r & 15;
        float em = MI[il] * MJ[jl];
        float y = sQ[il * 256 + c] * sK[jl * 256 + c] * (em * e1 + 1.f) +
                  em * e2;
        sHf[idx] = y;
        sNE[idx] = f2bf(YEA[c] + (YEM[c] + 1.f) * y);
      }
      __syncthreads();
      // --- online softmax: tile max first, single rescale, then 16 exps ---
#pragma unroll
      for (int kk = 0; kk < 2; ++kk) {
        int cc = cb_t + 16 * kk;
        int c = c0 + cc;
        int slot = nc * 2 + kk;
        float yv[16];
        float tm = -1e30f;
#pragma unroll
        for (int jl = 0; jl < 16; ++jl) {
          yv[jl] = sHf[(il_t * 16 + jl) * 32 + cc];
          tm = fmaxf(tm, yv[jl]);
        }
        float mo = smM[slot];
        float mn = fmaxf(mo, tm);
        float es = __expf(mo - mn);
        float s = smS[slot] * es;
        float a = smA[slot] * es;
#pragma unroll
        for (int jl = 0; jl < 16; ++jl) {
          float e = __expf(yv[jl] - mn);
          s += e;
          a += e * sV[jl * 256 + c];
        }
        smS[slot] = s; smA[slot] = a; smM[slot] = mn;
      }
      // --- e_out partial GEMM; per wave fixed nt -> B frag hoisted ---
      {
        int ntv = w >> 1;          // 0..3
        int mtb = (w & 1) * 8;
        bf16x16 bb = frag_ld(sWeout, ntv * 16, 256, c0);
#pragma unroll
        for (int qq = 0; qq < 8; ++qq) {
          bf16x16 a = frag_ld(sNE, (mtb + qq) * 16, 32, 0);
          WMMA_BF(eacc[qq], a, bb);
        }
      }
      __syncthreads();
    } // nc

    // stage e_out result
    {
      int ntv = w >> 1, mtb = (w & 1) * 8;
#pragma unroll
      for (int qq = 0; qq < 8; ++qq)
        cstore_f32(sScrF, (mtb + qq) * 16, ntv * 16, 64, eacc[qq]);
    }
    __syncthreads();

    // --- row phase 1: newE mask + residual + LN(normE1); Eln -> regs & sA ---
    {
      float em = MI[t >> 4] * MJ[t & 15];
      float su = 0.f, sq = 0.f;
#pragma unroll
      for (int k = 0; k < 64; ++k) {
        float x = eR[k] + (sScrF[t * 64 + k] + bEo[k]) * em;
        eR[k] = x; su += x; sq += x * x;
      }
      float mean = su * (1.f / 64.f);
      float rstd = rsqrtf(sq * (1.f / 64.f) - mean * mean + EPS_);
#pragma unroll
      for (int k = 0; k < 64; ++k)
        eR[k] = (eR[k] - mean) * rstd * L1w[k] + L1b[k];
#pragma unroll
      for (int m2 = 0; m2 < 32; ++m2)
        sA32[t * 32 + m2] = packbf(eR[2 * m2], eR[2 * m2 + 1]);
    }
    __syncthreads();

    // --- E FFN: 64 -> 128 (relu) -> 64, hidden chunked at 64 ---
    floatx8 facc[8];
#pragma unroll
    for (int q = 0; q < 8; ++q)
      facc[q] = (floatx8){0.f, 0.f, 0.f, 0.f, 0.f, 0.f, 0.f, 0.f};
    {
      int ntv = w >> 1, mtb = (w & 1) * 8;
#pragma unroll
      for (int hc = 0; hc < 2; ++hc) {
        {
          bf16x16 b0 = frag_ld(sW1, hc * 64 + ntv * 16, 64, 0);
          bf16x16 b1 = frag_ld(sW1, hc * 64 + ntv * 16, 64, 32);
#pragma unroll
          for (int qq = 0; qq < 8; ++qq) {
            int mt = mtb + qq;
            bf16x16 a0 = frag_ld(sA, mt * 16, 64, 0);
            bf16x16 a1 = frag_ld(sA, mt * 16, 64, 32);
            floatx8 acc = {0.f, 0.f, 0.f, 0.f, 0.f, 0.f, 0.f, 0.f};
            WMMA_BF(acc, a0, b0);
            WMMA_BF(acc, a1, b1);
            cstore_relu_bf(sHu, mt * 16, ntv * 16, 64, acc, B1 + hc * 64);
          }
        }
        __syncthreads();
        {
          bf16x16 b0 = frag_ld(sW2, ntv * 16, 128, hc * 64);
          bf16x16 b1 = frag_ld(sW2, ntv * 16, 128, hc * 64 + 32);
#pragma unroll
          for (int qq = 0; qq < 8; ++qq) {
            int mt = mtb + qq;
            bf16x16 a0 = frag_ld(sHu, mt * 16, 64, 0);
            bf16x16 a1 = frag_ld(sHu, mt * 16, 64, 32);
            WMMA_BF(facc[qq], a0, b0);
            WMMA_BF(facc[qq], a1, b1);
          }
        }
        __syncthreads();
      }
#pragma unroll
      for (int qq = 0; qq < 8; ++qq)
        cstore_f32(sScrF, (mtb + qq) * 16, ntv * 16, 64, facc[qq]);
    }
    __syncthreads();

    // --- row phase 2: FFN residual + LN(normE2) -> global E out ---
    {
      float su = 0.f, sq = 0.f;
      float xv[64];
#pragma unroll
      for (int k = 0; k < 64; ++k) {
        float x = eR[k] + sScrF[t * 64 + k] + B2[k];
        xv[k] = x; su += x; sq += x * x;
      }
      float mean = su * (1.f / 64.f);
      float rstd = rsqrtf(sq * (1.f / 64.f) - mean * mean + EPS_);
      float4* od4 = (float4*)(Eout + ((size_t)(b * 128 + ri) * 128 + rj) * 64);
#pragma unroll
      for (int k4 = 0; k4 < 16; ++k4) {
        float4 o;
        o.x = (xv[4 * k4 + 0] - mean) * rstd * L2w[4 * k4 + 0] + L2b[4 * k4 + 0];
        o.y = (xv[4 * k4 + 1] - mean) * rstd * L2w[4 * k4 + 1] + L2b[4 * k4 + 1];
        o.z = (xv[4 * k4 + 2] - mean) * rstd * L2w[4 * k4 + 2] + L2b[4 * k4 + 2];
        o.w = (xv[4 * k4 + 3] - mean) * rstd * L2w[4 * k4 + 3] + L2b[4 * k4 + 3];
        od4[k4] = o;
      }
    }
  } // jtl

  // --- write partial softmax state for this j-half ---
  {
    int p = (b * 128 + i0 + il_t) * 2 + jh;
#pragma unroll
    for (int k = 0; k < 16; ++k) {
      int c = cb_t + 16 * k;
      wsM[(size_t)p * 256 + c] = smM[k];
      wsS[(size_t)p * 256 + c] = smS[k];
      wsA[(size_t)p * 256 + c] = smA[k];
    }
  }
}

// ---------------------------------------------------------------------------
// Kernel 3b: merge the two j-half softmax states; newX_pre = yxa + (yxm+1)*wV.
// Grid 1024 x 256 threads, one (row, channel) element each.
// ---------------------------------------------------------------------------
__global__ void k_attfin(const float* __restrict__ wsM,
                         const float* __restrict__ wsS,
                         const float* __restrict__ wsA,
                         const float* __restrict__ yxa,
                         const float* __restrict__ yxm,
                         float* __restrict__ nxpre) {
  int idx = blockIdx.x * 256 + threadIdx.x; // over 8*128*256
  int c = idx & 255;
  int row = idx >> 8; // b*128 + i
  int b = row >> 7;
  size_t p0 = (size_t)(row * 2) * 256 + c;
  size_t p1 = (size_t)(row * 2 + 1) * 256 + c;
  float m0 = wsM[p0], m1 = wsM[p1];
  float m = fmaxf(m0, m1);
  float e0 = __expf(m0 - m), e1 = __expf(m1 - m);
  float s = wsS[p0] * e0 + wsS[p1] * e1;
  float a = wsA[p0] * e0 + wsA[p1] * e1;
  float wv = a / s;
  nxpre[idx] = yxa[b * 256 + c] + (yxm[b * 256 + c] + 1.f) * wv;
}

// ---------------------------------------------------------------------------
// Kernel 4: X path: x_out GEMM + LN1, FFN 256->2048->256 + LN2.
// Grid 16 (64-row tiles), 256 threads.
// ---------------------------------------------------------------------------
#define XF_SA 0        // ushort [64][256]
#define XF_SB 32768    // ushort [128][256]  (aliased later as float sC2 [64][256])
#define XF_SW2 98304   // ushort [256][128]
#define XF_SH 163840   // ushort [64][128]
#define XF_SC 180224   // float  [64][256]
#define XF_TOTAL 245760

__global__ void __launch_bounds__(256)
k_xff(const float* __restrict__ X, const float* __restrict__ mask,
      const float* __restrict__ nxpre,
      const float* __restrict__ Wxo, const float* __restrict__ bxo,
      const float* __restrict__ W1, const float* __restrict__ b1,
      const float* __restrict__ W2, const float* __restrict__ b2,
      const float* __restrict__ ln1w, const float* __restrict__ ln1b,
      const float* __restrict__ ln2w, const float* __restrict__ ln2b,
      float* __restrict__ Xout) {
  extern __shared__ char sm[];
  unsigned short* sA = (unsigned short*)(sm + XF_SA);
  unsigned short* sB = (unsigned short*)(sm + XF_SB);
  unsigned short* sW2c = (unsigned short*)(sm + XF_SW2);
  unsigned short* sH = (unsigned short*)(sm + XF_SH);
  float* sC = (float*)(sm + XF_SC);
  float* sC2 = (float*)(sm + XF_SB); // reuse after FFN GEMMs
  int t = threadIdx.x, w = t >> 5;
  int rows0 = blockIdx.x * 64;

  for (int idx = t; idx < 16384; idx += 256) {
    int r = idx >> 8, k = idx & 255;
    sA[r * 256 + k] = f2bf(nxpre[(rows0 + r) * 256 + k]);
  }
  __syncthreads();
  // x_out GEMM (K=256, N=256 in two 128-col chunks); per wave nt = w.
  for (int nch = 0; nch < 2; ++nch) {
    int n0 = nch * 128;
    for (int idx = t; idx < 32768; idx += 256) {
      int n = idx & 127, k = idx >> 7;
      sB[n * 256 + k] = f2bf(Wxo[k * 256 + n0 + n]);
    }
    __syncthreads();
    {
      floatx8 acc[4];
#pragma unroll
      for (int mt = 0; mt < 4; ++mt)
        acc[mt] = (floatx8){0.f, 0.f, 0.f, 0.f, 0.f, 0.f, 0.f, 0.f};
#pragma unroll
      for (int ks = 0; ks < 8; ++ks) {
        bf16x16 bb = frag_ld(sB, w * 16, 256, ks * 32);
#pragma unroll
        for (int mt = 0; mt < 4; ++mt) {
          bf16x16 a = frag_ld(sA, mt * 16, 256, ks * 32);
          WMMA_BF(acc[mt], a, bb);
        }
      }
#pragma unroll
      for (int mt = 0; mt < 4; ++mt)
        cstore_f32(sC, mt * 16, n0 + w * 16, 256, acc[mt]);
    }
    __syncthreads();
  }
  // row phase 1: residual + LN(normX1); Xa -> sC (f32) and sA (bf16)
  if (t < 64) {
    int gr = rows0 + t;
    float mk = mask[gr];
    float su = 0.f, sq = 0.f;
    for (int c = 0; c < 256; ++c) {
      float tv = X[gr * 256 + c] + (sC[t * 256 + c] + bxo[c]) * mk;
      sC[t * 256 + c] = tv; su += tv; sq += tv * tv;
    }
    float mean = su * (1.f / 256.f);
    float rstd = rsqrtf(sq * (1.f / 256.f) - mean * mean + EPS_);
    for (int c = 0; c < 256; ++c) {
      float xa = (sC[t * 256 + c] - mean) * rstd * ln1w[c] + ln1b[c];
      sC[t * 256 + c] = xa;
      sA[t * 256 + c] = f2bf(xa);
    }
  }
  __syncthreads();
  // FFN 256 -> 2048 (relu) -> 256, hidden chunked at 128
  floatx8 facc[8];
#pragma unroll
  for (int q = 0; q < 8; ++q)
    facc[q] = (floatx8){0.f, 0.f, 0.f, 0.f, 0.f, 0.f, 0.f, 0.f};
  for (int hc = 0; hc < 16; ++hc) {
    int h0 = hc * 128;
    for (int idx = t; idx < 32768; idx += 256) {
      int n = idx & 127, k = idx >> 7;
      sB[n * 256 + k] = f2bf(W1[k * 2048 + h0 + n]);
    }
    for (int idx = t; idx < 32768; idx += 256) {
      int n = idx & 255, k = idx >> 8;
      sW2c[n * 128 + k] = f2bf(W2[(h0 + k) * 256 + n]);
    }
    __syncthreads();
    // GEMM1: per wave nt = w; B frag hoisted per ks.
    {
      floatx8 acc[4];
#pragma unroll
      for (int mt = 0; mt < 4; ++mt)
        acc[mt] = (floatx8){0.f, 0.f, 0.f, 0.f, 0.f, 0.f, 0.f, 0.f};
#pragma unroll
      for (int ks = 0; ks < 8; ++ks) {
        bf16x16 bb = frag_ld(sB, w * 16, 256, ks * 32);
#pragma unroll
        for (int mt = 0; mt < 4; ++mt) {
          bf16x16 a = frag_ld(sA, mt * 16, 256, ks * 32);
          WMMA_BF(acc[mt], a, bb);
        }
      }
#pragma unroll
      for (int mt = 0; mt < 4; ++mt)
        cstore_relu_bf(sH, mt * 16, w * 16, 128, acc[mt], b1 + h0);
    }
    __syncthreads();
    // GEMM2: per wave nt in {2w, 2w+1}; all B frags preloaded, A shared.
    {
      bf16x16 b0[4], b1f[4];
#pragma unroll
      for (int ks = 0; ks < 4; ++ks) {
        b0[ks] = frag_ld(sW2c, (w * 2) * 16, 128, ks * 32);
        b1f[ks] = frag_ld(sW2c, (w * 2 + 1) * 16, 128, ks * 32);
      }
#pragma unroll
      for (int mt = 0; mt < 4; ++mt) {
        bf16x16 a[4];
#pragma unroll
        for (int ks = 0; ks < 4; ++ks) a[ks] = frag_ld(sH, mt * 16, 128, ks * 32);
#pragma unroll
        for (int ks = 0; ks < 4; ++ks) WMMA_BF(facc[mt], a[ks], b0[ks]);
#pragma unroll
        for (int ks = 0; ks < 4; ++ks) WMMA_BF(facc[4 + mt], a[ks], b1f[ks]);
      }
    }
    __syncthreads();
  }
#pragma unroll
  for (int q = 0; q < 8; ++q) {
    int job = w * 8 + q;
    cstore_f32(sC2, (job & 3) * 16, (job >> 2) * 16, 256, facc[q]);
  }
  __syncthreads();
  if (t < 64) {
    int gr = rows0 + t;
    float su = 0.f, sq = 0.f;
    for (int c = 0; c < 256; ++c) {
      float tv = sC[t * 256 + c] + sC2[t * 256 + c] + b2[c];
      sC2[t * 256 + c] = tv; su += tv; sq += tv * tv;
    }
    float mean = su * (1.f / 256.f);
    float rstd = rsqrtf(sq * (1.f / 256.f) - mean * mean + EPS_);
    for (int c = 0; c < 256; ++c)
      Xout[gr * 256 + c] = (sC2[t * 256 + c] - mean) * rstd * ln2w[c] + ln2b[c];
  }
}

// ---------------------------------------------------------------------------
// Stats kernels for the y path.
// ---------------------------------------------------------------------------
__global__ void k_xstats(const float* __restrict__ X, float* __restrict__ zx) {
  int b = blockIdx.x, c = threadIdx.x;
  float s = 0.f, sq = 0.f, mn = 1e30f, mx = -1e30f;
  for (int i = 0; i < 128; ++i) {
    float v = X[(b * 128 + i) * 256 + c];
    s += v; sq += v * v; mn = fminf(mn, v); mx = fmaxf(mx, v);
  }
  float mean = s / 128.f;
  float sd = sqrtf(fmaxf((sq - 128.f * mean * mean) / 127.f, 0.f));
  zx[b * 1024 + c] = mean;
  zx[b * 1024 + 256 + c] = mn;
  zx[b * 1024 + 512 + c] = mx;
  zx[b * 1024 + 768 + c] = sd;
}

__global__ void k_estats1(const float* __restrict__ E, float* __restrict__ part) {
  __shared__ float rs[256], rmn[256], rmx[256], rsq[256];
  int b = blockIdx.y, s = blockIdx.x, t = threadIdx.x;
  int c = t & 63, sub = t >> 6;
  int p0 = s * 512 + sub * 128;
  float su = 0.f, sq = 0.f, mn = 1e30f, mx = -1e30f;
  for (int q = 0; q < 128; ++q) {
    float v = E[(size_t)(b * 16384 + p0 + q) * 64 + c];
    su += v; sq += v * v; mn = fminf(mn, v); mx = fmaxf(mx, v);
  }
  rs[t] = su; rsq[t] = sq; rmn[t] = mn; rmx[t] = mx;
  __syncthreads();
  if (t < 64) {
    for (int u = 1; u < 4; ++u) {
      su += rs[u * 64 + t]; sq += rsq[u * 64 + t];
      mn = fminf(mn, rmn[u * 64 + t]); mx = fmaxf(mx, rmx[u * 64 + t]);
    }
    float* o = part + ((b * 32 + s) * 64 + t) * 4;
    o[0] = su; o[1] = mn; o[2] = mx; o[3] = sq;
  }
}

__global__ void k_estats2(const float* __restrict__ part, float* __restrict__ ze) {
  int b = blockIdx.x, c = threadIdx.x; // 64 threads
  float su = 0.f, sq = 0.f, mn = 1e30f, mx = -1e30f;
  for (int s = 0; s < 32; ++s) {
    const float* o = part + ((b * 32 + s) * 64 + c) * 4;
    su += o[0]; mn = fminf(mn, o[1]); mx = fmaxf(mx, o[2]); sq += o[3];
  }
  float n = 16384.f;
  float mean = su / n;
  float sd = sqrtf(fmaxf((sq - n * mean * mean) / (n - 1.f), 0.f));
  ze[b * 256 + c] = mean;
  ze[b * 256 + 64 + c] = mn;
  ze[b * 256 + 128 + c] = mx;
  ze[b * 256 + 192 + c] = sd;
}

// ---------------------------------------------------------------------------
// Kernel: full y path (tiny). Grid 8, 64 threads.
// ---------------------------------------------------------------------------
__global__ void k_ycomb(const float* __restrict__ y, const float* __restrict__ zx,
                        const float* __restrict__ ze,
                        const float* Wxy, const float* bxy,
                        const float* Wey, const float* bey,
                        const float* Wo1, const float* bo1,
                        const float* Wo2, const float* bo2,
                        const float* ln1w, const float* ln1b,
                        const float* Wl1, const float* bl1,
                        const float* Wl2, const float* bl2,
                        const float* ln2w, const float* ln2b,
                        float* __restrict__ yout) {
  __shared__ float sz[1024], se[256], sy[64], t0[64], t1[64], h[2048];
  int b = blockIdx.x, c = threadIdx.x;
  for (int i = c; i < 1024; i += 64) sz[i] = zx[b * 1024 + i];
  for (int i = c; i < 256; i += 64) se[i] = ze[b * 256 + i];
  sy[c] = y[b * 64 + c];
  __syncthreads();
  float xy = bxy[c];
  for (int k = 0; k < 1024; ++k) xy += sz[k] * Wxy[k * 64 + c];
  float ey = bey[c];
  for (int k = 0; k < 256; ++k) ey += se[k] * Wey[k * 64 + c];
  t0[c] = sy[c] + xy + ey;
  __syncthreads();
  float hh = bo1[c];
  for (int k = 0; k < 64; ++k) hh += t0[k] * Wo1[k * 64 + c];
  t1[c] = fmaxf(hh, 0.f);
  __syncthreads();
  float o = bo2[c];
  for (int k = 0; k < 64; ++k) o += t1[k] * Wo2[k * 64 + c];
  t0[c] = sy[c] + o; // y + new_y
  __syncthreads();
  float m = 0.f;
  for (int k = 0; k < 64; ++k) m += t0[k];
  m *= (1.f / 64.f);
  float v = 0.f;
  for (int k = 0; k < 64; ++k) { float d = t0[k] - m; v += d * d; }
  float rstd = rsqrtf(v * (1.f / 64.f) + EPS_);
  float ya = (t0[c] - m) * rstd * ln1w[c] + ln1b[c];
  t1[c] = ya;
  __syncthreads();
  for (int j = c * 32; j < c * 32 + 32; ++j) {
    float s = bl1[j];
    for (int k = 0; k < 64; ++k) s += t1[k] * Wl1[k * 2048 + j];
    h[j] = fmaxf(s, 0.f);
  }
  __syncthreads();
  float o2 = bl2[c];
  for (int j = 0; j < 2048; ++j) o2 += h[j] * Wl2[j * 64 + c];
  t0[c] = ya + o2;
  __syncthreads();
  m = 0.f;
  for (int k = 0; k < 64; ++k) m += t0[k];
  m *= (1.f / 64.f);
  v = 0.f;
  for (int k = 0; k < 64; ++k) { float d = t0[k] - m; v += d * d; }
  rstd = rsqrtf(v * (1.f / 64.f) + EPS_);
  yout[b * 64 + c] = (t0[c] - m) * rstd * ln2w[c] + ln2b[c];
}

// ---------------------------------------------------------------------------
extern "C" void kernel_launch(void* const* d_in, const int* in_sizes, int n_in,
                              void* d_out, int out_size, void* d_ws,
                              size_t ws_size, hipStream_t stream) {
  (void)in_sizes; (void)n_in; (void)out_size; (void)ws_size;
  const float* X = (const float*)d_in[0];
  const float* E = (const float*)d_in[1];
  const float* y = (const float*)d_in[2];
  const float* mask = (const float*)d_in[3];
  // params: JAX pytree dict order (keys sorted), (w, b) leaves per key.
  int i = 4;
  const float* eadd_w = (const float*)d_in[i++]; const float* eadd_b = (const float*)d_in[i++];
  const float* emul_w = (const float*)d_in[i++]; const float* emul_b = (const float*)d_in[i++];
  const float* eout_w = (const float*)d_in[i++]; const float* eout_b = (const float*)d_in[i++];
  const float* ey_w   = (const float*)d_in[i++]; const float* ey_b   = (const float*)d_in[i++];
  const float* k_w    = (const float*)d_in[i++]; const float* k_b    = (const float*)d_in[i++];
  const float* linE1_w= (const float*)d_in[i++]; const float* linE1_b= (const float*)d_in[i++];
  const float* linE2_w= (const float*)d_in[i++]; const float* linE2_b= (const float*)d_in[i++];
  const float* linX1_w= (const float*)d_in[i++]; const float* linX1_b= (const float*)d_in[i++];
  const float* linX2_w= (const float*)d_in[i++]; const float* linX2_b= (const float*)d_in[i++];
  const float* liny1_w= (const float*)d_in[i++]; const float* liny1_b= (const float*)d_in[i++];
  const float* liny2_w= (const float*)d_in[i++]; const float* liny2_b= (const float*)d_in[i++];
  const float* nE1_w  = (const float*)d_in[i++]; const float* nE1_b  = (const float*)d_in[i++];
  const float* nE2_w  = (const float*)d_in[i++]; const float* nE2_b  = (const float*)d_in[i++];
  const float* nX1_w  = (const float*)d_in[i++]; const float* nX1_b  = (const float*)d_in[i++];
  const float* nX2_w  = (const float*)d_in[i++]; const float* nX2_b  = (const float*)d_in[i++];
  const float* ny1_w  = (const float*)d_in[i++]; const float* ny1_b  = (const float*)d_in[i++];
  const float* ny2_w  = (const float*)d_in[i++]; const float* ny2_b  = (const float*)d_in[i++];
  const float* q_w    = (const float*)d_in[i++]; const float* q_b    = (const float*)d_in[i++];
  const float* v_w    = (const float*)d_in[i++]; const float* v_b    = (const float*)d_in[i++];
  const float* xout_w = (const float*)d_in[i++]; const float* xout_b = (const float*)d_in[i++];
  const float* xy_w   = (const float*)d_in[i++]; const float* xy_b   = (const float*)d_in[i++];
  const float* yea_w  = (const float*)d_in[i++]; const float* yea_b  = (const float*)d_in[i++];
  const float* yem_w  = (const float*)d_in[i++]; const float* yem_b  = (const float*)d_in[i++];
  const float* yo1_w  = (const float*)d_in[i++]; const float* yo1_b  = (const float*)d_in[i++];
  const float* yo2_w  = (const float*)d_in[i++]; const float* yo2_b  = (const float*)d_in[i++];
  const float* yxa_w  = (const float*)d_in[i++]; const float* yxa_b  = (const float*)d_in[i++];
  const float* yxm_w  = (const float*)d_in[i++]; const float* yxm_b  = (const float*)d_in[i++];
  // y_y is unused by the forward pass.

  float* ws = (float*)d_ws;
  float* wQ = ws;
  float* wK = ws + 262144;
  float* wV = ws + 524288;
  float* wYEA = ws + 786432;
  float* wYEM = wYEA + 2048;
  float* wYXA = wYEM + 2048;
  float* wYXM = wYXA + 2048;
  float* wNX = wYXM + 2048;   // 262144
  float* wZX = wNX + 262144;  // 8192
  float* wEP = wZX + 8192;    // 65536
  float* wZE = wEP + 65536;   // 2048
  float* wSM = wZE + 2048;    // 524288 (2048 x 256) partial softmax max
  float* wSS = wSM + 524288;  // 524288 partial softmax sum
  float* wSA = wSS + 524288;  // 524288 partial weighted-V accum

  float* oX = (float*)d_out;
  float* oE = oX + 262144;
  float* oY = oE + 8388608;

  k_xproj<<<dim3(16), dim3(256), 82176, stream>>>(
      X, mask, q_w, q_b, k_w, k_b, v_w, v_b, wQ, wK, wV);
  k_ymats<<<dim3(8), dim3(256), 0, stream>>>(
      y, yea_w, yea_b, yem_w, yem_b, yxa_w, yxa_b, yxm_w, yxm_b,
      wYEA, wYEM, wYXA, wYXM);
  k_xstats<<<dim3(8), dim3(256), 0, stream>>>(X, wZX);
  k_estats1<<<dim3(32, 8), dim3(256), 0, stream>>>(E, wEP);
  k_estats2<<<dim3(8), dim3(64), 0, stream>>>(wEP, wZE);
  k_emain<<<dim3(8, 8, 2), dim3(256), EM_TOTAL, stream>>>(
      E, mask, wQ, wK, wV, emul_w, emul_b, eadd_w, eadd_b, eout_w, eout_b,
      linE1_w, linE1_b, linE2_w, linE2_b, nE1_w, nE1_b, nE2_w, nE2_b,
      wYEA, wYEM, oE, wSM, wSS, wSA);
  k_attfin<<<dim3(1024), dim3(256), 0, stream>>>(
      wSM, wSS, wSA, wYXA, wYXM, wNX);
  k_xff<<<dim3(16), dim3(256), XF_TOTAL, stream>>>(
      X, mask, wNX, xout_w, xout_b, linX1_w, linX1_b, linX2_w, linX2_b,
      nX1_w, nX1_b, nX2_w, nX2_b, oX);
  k_ycomb<<<dim3(8), dim3(64), 0, stream>>>(
      y, wZX, wZE, xy_w, xy_b, ey_w, ey_b, yo1_w, yo1_b, yo2_w, yo2_b,
      ny1_w, ny1_b, liny1_w, liny1_b, liny2_w, liny2_b, ny2_w, ny2_b, oY);
}
